// DLinear_2705829396667
// MI455X (gfx1250) — compile-verified, matrix operand order checked by reading, not run on hardware
//
#include <hip/hip_runtime.h>

#define B_DIM 4096
#define S_DIM 2048
#define O_DIM 720
#define O_PAD 768
#define WIN   25

#define BM 128
#define BN 64
#define BK 32
#define NTHREADS 128            // 4 waves; each wave owns M=32 x N=64
#define XSTRIDE 36              // padded f32 row stride in LDS (16B aligned, conflict-free)
#define WSTRIDE 40              // padded bf16 row stride in LDS
#define KSTEPS (S_DIM / BK)

typedef __attribute__((ext_vector_type(16))) __bf16 v16bf;
typedef __attribute__((ext_vector_type(8)))  __bf16 v8bf;
typedef __attribute__((ext_vector_type(8)))  float  v8f;
typedef __attribute__((ext_vector_type(4)))  float  v4f;

// CDNA5 async global->LDS staging (ASYNCcnt path), with sync fallback.
#if defined(__HIP_DEVICE_COMPILE__) && __has_builtin(__builtin_amdgcn_global_load_async_to_lds_b128)
#define USE_ASYNC_LDS 1
#else
#define USE_ASYNC_LDS 0
#endif

#if USE_ASYNC_LDS
typedef int v4i_ __attribute__((vector_size(16)));
typedef __attribute__((address_space(1))) v4i_ gv4i;   // global int4
typedef __attribute__((address_space(3))) v4i_ lv4i;   // LDS int4

__device__ __forceinline__ void async_cp16(const void* g, void* l) {
    __builtin_amdgcn_global_load_async_to_lds_b128(
        (gv4i*)g, (lv4i*)l, /*imm offset*/0, /*cpol*/0);
}
__device__ __forceinline__ void wait_async0() {
#if __has_builtin(__builtin_amdgcn_s_wait_asynccnt)
    __builtin_amdgcn_s_wait_asynccnt(0);
#else
    asm volatile("s_wait_asynccnt 0" ::: "memory");
#endif
}
#endif

// ---------------------------------------------------------------------------
// Stage 1: fold the causal-window mean into the weights.
//   W_eff[o,t] = sW[o,t] + sum_{s=t}^{min(S-1,t+24)} (tW[o,s]-sW[o,s]) / min(s+1,25)
// Split W_eff into bf16 hi/lo for the bf16x3 WMMA GEMM; pad O to 768 with zeros.
// Also b_eff = tb + sb.
// ---------------------------------------------------------------------------
__global__ __launch_bounds__(256)
void dlinear_prep(const float* __restrict__ tW, const float* __restrict__ tb,
                  const float* __restrict__ sW, const float* __restrict__ sb,
                  __bf16* __restrict__ whi, __bf16* __restrict__ wlo,
                  float* __restrict__ beff)
{
    const int idx = blockIdx.x * 256 + threadIdx.x;   // 0 .. O_PAD*S_DIM-1
    const int o = idx >> 11;          // / 2048
    const int t = idx & (S_DIM - 1);
    float w = 0.0f;
    if (o < O_DIM) {
        const float* tr = tW + o * S_DIM;
        const float* se = sW + o * S_DIM;
        float acc = 0.0f;
        int send = t + WIN; if (send > S_DIM) send = S_DIM;
        for (int s = t; s < send; ++s) {
            float cnt = (float)((s + 1 < WIN) ? (s + 1) : WIN);
            acc += (tr[s] - se[s]) / cnt;
        }
        w = se[t] + acc;
    }
    __bf16 hi = (__bf16)w;
    whi[idx] = hi;
    wlo[idx] = (__bf16)(w - (float)hi);
    if (idx < O_PAD)
        beff[idx] = (idx < O_DIM) ? (tb[idx] + sb[idx]) : 0.0f;
}

// ---------------------------------------------------------------------------
// Stage 2: out[b,o] = sum_s x[b,s] * W_eff[o,s] + b_eff[o]
// bf16x3 emulated-fp32 GEMM via v_wmma_f32_16x16x32_bf16.
// 4 waves/block; each wave: 32x64 output (2x4 WMMA tiles) -> 24 WMMA/K-step
// amortizing the per-wave A-fragment f32->bf16 split over the full N extent.
// LDS double buffered; tiles staged with async global->LDS copies (ASYNCcnt).
// ---------------------------------------------------------------------------
__global__ __launch_bounds__(NTHREADS)
void dlinear_gemm(const float* __restrict__ x,
                  const __bf16* __restrict__ whi,
                  const __bf16* __restrict__ wlo,
                  const float* __restrict__ beff,
                  float* __restrict__ out)
{
    __shared__ float  xs[2][BM * XSTRIDE];       // 36 KB : x tile (raw f32)
    __shared__ __bf16 bhiS[2][BN * WSTRIDE];     // 10 KB : W_hi tile
    __shared__ __bf16 bloS[2][BN * WSTRIDE];     // 10 KB : W_lo tile

    const int tid   = threadIdx.x;
    const int lane  = tid & 31;
    const int waveM = tid >> 5;    // 4 waves in M; each wave spans all of BN
    const int hlf   = lane >> 4;   // upper/lower 16-lane half
    const int l16   = lane & 15;

    const int rowX0 = blockIdx.y * BM;   // batch-row base
    const int rowW0 = blockIdx.x * BN;   // output-feature base (padded space)

    // staging map (16B per thread per transfer), 128 threads
    const int xRow = tid >> 3;           // + 16*i, i=0..7
    const int xCol = (tid & 7) * 4;
    const int wRow = tid >> 2;           // + 32*i, i=0..1
    const int wCol = (tid & 3) * 8;

#if USE_ASYNC_LDS
    auto stage_tiles = [&](int buf, int k0) {
#pragma unroll
        for (int i = 0; i < 8; ++i)
            async_cp16(x + (rowX0 + xRow + 16 * i) * S_DIM + k0 + xCol,
                       &xs[buf][(xRow + 16 * i) * XSTRIDE + xCol]);
#pragma unroll
        for (int i = 0; i < 2; ++i) {
            const int wOff = (rowW0 + wRow + 32 * i) * S_DIM + k0 + wCol;
            const int lOff = (wRow + 32 * i) * WSTRIDE + wCol;
            async_cp16(whi + wOff, &bhiS[buf][lOff]);
            async_cp16(wlo + wOff, &bloS[buf][lOff]);
        }
    };
#else
    v4f  xr[8];
    v8bf wh[2], wl[2];
    auto fetch = [&](int k0) {
        const float* xp = x + (rowX0 + xRow) * S_DIM + k0 + xCol;
#pragma unroll
        for (int i = 0; i < 8; ++i)
            xr[i] = *(const v4f*)(xp + i * 16 * S_DIM);
#pragma unroll
        for (int i = 0; i < 2; ++i) {
            const int wOff = (rowW0 + wRow + 32 * i) * S_DIM + k0 + wCol;
            wh[i] = *(const v8bf*)(whi + wOff);
            wl[i] = *(const v8bf*)(wlo + wOff);
        }
    };
    auto stage = [&](int buf) {
#pragma unroll
        for (int i = 0; i < 8; ++i)
            *(v4f*)&xs[buf][(xRow + 16 * i) * XSTRIDE + xCol] = xr[i];
#pragma unroll
        for (int i = 0; i < 2; ++i) {
            const int lOff = (wRow + 32 * i) * WSTRIDE + wCol;
            *(v8bf*)&bhiS[buf][lOff] = wh[i];
            *(v8bf*)&bloS[buf][lOff] = wl[i];
        }
    };
#endif

    v8f c[2][4];
#pragma unroll
    for (int mt = 0; mt < 2; ++mt)
#pragma unroll
        for (int nt = 0; nt < 4; ++nt)
#pragma unroll
            for (int r = 0; r < 8; ++r)
                c[mt][nt][r] = 0.0f;

    auto compute = [&](int buf) {
        v16bf ahi[2], alo[2], bh[4], bl[4];
        // A fragments: convert f32 -> bf16 hi/lo at load time.
        // Lane layout (ISA 7.12.2): e0..7 = K(hlf*8 ..), e8..15 = K(16+hlf*8 ..)
#pragma unroll
        for (int mt = 0; mt < 2; ++mt) {
            const float* p = &xs[buf][(waveM * 32 + mt * 16 + l16) * XSTRIDE + hlf * 8];
            v4f f0 = *(const v4f*)(p);
            v4f f1 = *(const v4f*)(p + 4);
            v4f f2 = *(const v4f*)(p + 16);
            v4f f3 = *(const v4f*)(p + 20);
            float fv[16];
#pragma unroll
            for (int i = 0; i < 4; ++i) {
                fv[i] = f0[i]; fv[4 + i] = f1[i]; fv[8 + i] = f2[i]; fv[12 + i] = f3[i];
            }
#pragma unroll
            for (int i = 0; i < 16; ++i) {
                __bf16 h = (__bf16)fv[i];
                ahi[mt][i] = h;
                alo[mt][i] = (__bf16)(fv[i] - (float)h);
            }
        }
        // B fragments: W rows are K-contiguous (N-major), same per-lane K split.
#pragma unroll
        for (int nt = 0; nt < 4; ++nt) {
            const int ro = (nt * 16 + l16) * WSTRIDE + hlf * 8;
            v8bf h0 = *(const v8bf*)&bhiS[buf][ro];
            v8bf h1 = *(const v8bf*)&bhiS[buf][ro + 16];
            v8bf l0 = *(const v8bf*)&bloS[buf][ro];
            v8bf l1 = *(const v8bf*)&bloS[buf][ro + 16];
#pragma unroll
            for (int i = 0; i < 8; ++i) {
                bh[nt][i] = h0[i]; bh[nt][8 + i] = h1[i];
                bl[nt][i] = l0[i]; bl[nt][8 + i] = l1[i];
            }
        }
        // bf16x3: hi*hi, hi*lo, lo*hi — each accumulator reused only every
        // 8 WMMAs (hides the WMMA->WMMA RAW latency).
#pragma unroll
        for (int mt = 0; mt < 2; ++mt)
#pragma unroll
            for (int nt = 0; nt < 4; ++nt)
                c[mt][nt] = __builtin_amdgcn_wmma_f32_16x16x32_bf16(
                    false, ahi[mt], false, bh[nt], (short)0, c[mt][nt], false, false);
#pragma unroll
        for (int mt = 0; mt < 2; ++mt)
#pragma unroll
            for (int nt = 0; nt < 4; ++nt)
                c[mt][nt] = __builtin_amdgcn_wmma_f32_16x16x32_bf16(
                    false, ahi[mt], false, bl[nt], (short)0, c[mt][nt], false, false);
#pragma unroll
        for (int mt = 0; mt < 2; ++mt)
#pragma unroll
            for (int nt = 0; nt < 4; ++nt)
                c[mt][nt] = __builtin_amdgcn_wmma_f32_16x16x32_bf16(
                    false, alo[mt], false, bh[nt], (short)0, c[mt][nt], false, false);
    };

#if USE_ASYNC_LDS
    stage_tiles(0, 0);
    wait_async0();
    __syncthreads();
    int cur = 0;
    for (int kk = 0; kk < KSTEPS - 1; ++kk) {
        stage_tiles(cur ^ 1, (kk + 1) * BK);   // async copy overlaps WMMA below
        compute(cur);
        wait_async0();                          // own async writes to LDS done
        __syncthreads();                        // everyone done reading cur + staging
        cur ^= 1;
    }
    compute(cur);
#else
    fetch(0);
    stage(0);
    __syncthreads();
    int cur = 0;
    for (int kk = 0; kk < KSTEPS - 1; ++kk) {
        fetch((kk + 1) * BK);                   // overlap VMEM with WMMA
        compute(cur);
        stage(cur ^ 1);
        __syncthreads();
        cur ^= 1;
    }
    compute(cur);
#endif

    // Epilogue. C layout (ISA 7.12.2): M = vgpr + 8*hlf, N = l16.
#pragma unroll
    for (int nt = 0; nt < 4; ++nt) {
        int ob = rowW0 + nt * 16 + l16;
        if (ob < O_DIM) {
            float bias = beff[ob];
#pragma unroll
            for (int mt = 0; mt < 2; ++mt) {
                int rb = rowX0 + waveM * 32 + mt * 16 + hlf * 8;
#pragma unroll
                for (int r = 0; r < 8; ++r)
                    out[(rb + r) * O_DIM + ob] = c[mt][nt][r] + bias;
            }
        }
    }
}

// ---------------------------------------------------------------------------
extern "C" void kernel_launch(void* const* d_in, const int* in_sizes, int n_in,
                              void* d_out, int out_size, void* d_ws, size_t ws_size,
                              hipStream_t stream) {
    (void)in_sizes; (void)n_in; (void)out_size; (void)ws_size;
    const float* x  = (const float*)d_in[0];
    const float* tW = (const float*)d_in[1];
    const float* tb = (const float*)d_in[2];
    const float* sW = (const float*)d_in[3];
    const float* sb = (const float*)d_in[4];
    float* out = (float*)d_out;

    char* ws = (char*)d_ws;
    __bf16* whi  = (__bf16*)ws;                                      // 768*2048*2 B
    __bf16* wlo  = (__bf16*)(ws + (size_t)O_PAD * S_DIM * 2);        // 768*2048*2 B
    float*  beff = (float*)(ws + (size_t)O_PAD * S_DIM * 4);         // 768*4 B

    dlinear_prep<<<(O_PAD * S_DIM) / 256, 256, 0, stream>>>(tW, tb, sW, sb, whi, wlo, beff);

    dim3 grid(O_PAD / BN, B_DIM / BM);   // 12 x 32 blocks, 128 threads each
    dlinear_gemm<<<grid, NTHREADS, 0, stream>>>(x, whi, wlo, beff, out);
}
